// Attention_57612691309274
// MI455X (gfx1250) — compile-verified, matrix operand order checked by reading, not run on hardware
//
#include <hip/hip_runtime.h>
#include <hip/hip_bf16.h>

typedef __attribute__((ext_vector_type(16))) __bf16 v16bf;
typedef __attribute__((ext_vector_type(8)))  float  v8f;

#define D_MODEL 1024
#define N_HEADS 16
#define D_HEAD  64
#define SEQ     2048
#define BATCH   2
#define MROWS   (BATCH*SEQ)          /* 4096 */
#define QKV_N   (3*N_HEADS*D_HEAD)   /* 3072 */

#define WMMA_BF16(a,b,c) \
  __builtin_amdgcn_wmma_f32_16x16x32_bf16(false,(a),false,(b),(short)0,(c),false,false)

// async global->LDS copy of 16 bytes (per lane); inst offset applies to both sides
#define ASYNC_LDS_B128(ldsoff, gptr) \
  asm volatile("global_load_async_to_lds_b128 %0, %1, off" :: "v"(ldsoff), "v"(gptr) : "memory")
#define ASYNC_LDS_B128_OFF16(ldsoff, gptr) \
  asm volatile("global_load_async_to_lds_b128 %0, %1, off offset:16" :: "v"(ldsoff), "v"(gptr) : "memory")
#define WAIT_ASYNC0() asm volatile("s_wait_asynccnt 0x0" ::: "memory")

__device__ __forceinline__ unsigned lds_off_of(const void* p) {
  // generic shared pointer: addr[31:0] is the LDS offset (aperture in high bits)
  return (unsigned)(uintptr_t)p;
}

// ---------------------------------------------------------------- pack kernels
__global__ void pack_x_kernel(const float* __restrict__ x, __bf16* __restrict__ xb, int n) {
  int i = blockIdx.x * blockDim.x + threadIdx.x;
  if (i < n) xb[i] = (__bf16)x[i];
}

// Wqkvt[c][d] = W_{mm}[h][d][e]  with c = mm*1024 + h*64 + e  (K-dim d contiguous)
__global__ void pack_wqkv_kernel(const float* __restrict__ wq, const float* __restrict__ wk,
                                 const float* __restrict__ wv, __bf16* __restrict__ wt) {
  int i = blockIdx.x * blockDim.x + threadIdx.x;
  if (i >= QKV_N * D_MODEL) return;
  int c = i / D_MODEL, d = i % D_MODEL;
  int mm = c / (N_HEADS * D_HEAD), rem = c % (N_HEADS * D_HEAD);
  int h = rem / D_HEAD, e = rem % D_HEAD;
  const float* src = (mm == 0) ? wq : (mm == 1) ? wk : wv;
  wt[i] = (__bf16)src[((size_t)h * D_MODEL + d) * D_HEAD + e];
}

// WOt[dm][he] = W_O[he][dm]  (plain 1024x1024 transpose)
__global__ void pack_wo_kernel(const float* __restrict__ wo, __bf16* __restrict__ wot) {
  int i = blockIdx.x * blockDim.x + threadIdx.x;
  if (i >= D_MODEL * D_MODEL) return;
  int dm = i / D_MODEL, he = i % D_MODEL;
  wot[i] = (__bf16)wo[(size_t)he * D_MODEL + dm];
}

__global__ void pack_bias_kernel(const float* __restrict__ bq, const float* __restrict__ bk,
                                 const float* __restrict__ bv, float* __restrict__ bias) {
  int i = blockIdx.x * blockDim.x + threadIdx.x;
  if (i >= QKV_N) return;
  int mm = i / 1024, r = i % 1024;
  bias[i] = (mm == 0) ? bq[r] : (mm == 1) ? bk[r] : bv[r];
}

// ---------------------------------------------------------------- GEMM kernel
// C[M,N] = (A16[M,K] @ Bt16[N,K]^T + bias[n]) * scale ; out as bf16 or f32.
// Block: 128(M) x 64(N), 8 waves. Double-buffered async global->LDS staging:
// the DMA for K-step i+1 is in flight while the WMMAs for K-step i execute.
__global__ __launch_bounds__(256) void gemm_bf16_kernel(
    const __bf16* __restrict__ A, const __bf16* __restrict__ Bt,
    const float* __restrict__ bias, float scale, int K, int ldc,
    __bf16* __restrict__ outb, float* __restrict__ outf)
{
  __shared__ __bf16 As[2][128 * 32];
  __shared__ __bf16 Bs[2][64 * 32];
  const int t = threadIdx.x;
  const int wave = t >> 5, lane = t & 31;
  const int lrow = lane & 15, half = lane >> 4;
  const int m0 = blockIdx.x * 128;
  const int n0 = blockIdx.y * 64;
  const int arow = t >> 1, ahalf = t & 1;   // 128 rows x 2 halves (32 B each)
  const int brow = t >> 2, bq = t & 3;      // 64 rows x 4 chunks (16 B each)

  unsigned dstA[2], dstB[2];
  dstA[0] = lds_off_of(&As[0][arow * 32 + ahalf * 16]);
  dstA[1] = lds_off_of(&As[1][arow * 32 + ahalf * 16]);
  dstB[0] = lds_off_of(&Bs[0][brow * 32 + bq * 8]);
  dstB[1] = lds_off_of(&Bs[1][brow * 32 + bq * 8]);

  const __bf16* srcA0 = A + (size_t)(m0 + arow) * K + ahalf * 16;
  const __bf16* srcB0 = Bt + (size_t)(n0 + brow) * K + bq * 8;

  // prologue: stage K-step 0 into buffer 0
  ASYNC_LDS_B128(dstA[0], srcA0);
  ASYNC_LDS_B128_OFF16(dstA[0], srcA0);
  ASYNC_LDS_B128(dstB[0], srcB0);

  v8f acc[4] = {};
  int buf = 0;
  for (int kt = 0; kt < K; kt += 32, buf ^= 1) {
    WAIT_ASYNC0();        // own writes of the current buffer complete
    __syncthreads();      // everyone's writes of current + reads of other done
    if (kt + 32 < K) {    // kick off DMA for the next K-step (overlaps compute)
      const __bf16* sa = srcA0 + kt + 32;
      const __bf16* sb = srcB0 + kt + 32;
      ASYNC_LDS_B128(dstA[buf ^ 1], sa);
      ASYNC_LDS_B128_OFF16(dstA[buf ^ 1], sa);
      ASYNC_LDS_B128(dstB[buf ^ 1], sb);
      if (kt + 64 < K) {
        __builtin_prefetch(sa + 32, 0, 1);
        __builtin_prefetch(sb + 32, 0, 1);
      }
    }
    const __bf16* Ab = &As[buf][0];
    const __bf16* Bb = &Bs[buf][0];
    // A fragment: row = lane%16 (of this wave's 16 rows), k = (lane/16)*16 .. +15
    v16bf afrag = *(const v16bf*)(Ab + (wave * 16 + lrow) * 32 + half * 16);
    v16bf bfrag[4];
#pragma unroll
    for (int j = 0; j < 4; ++j)  // B fragment: col = j*16+lane%16, k contiguous
      bfrag[j] = *(const v16bf*)(Bb + (j * 16 + lrow) * 32 + half * 16);
#pragma unroll
    for (int j = 0; j < 4; ++j)
      acc[j] = WMMA_BF16(afrag, bfrag[j], acc[j]);
  }

#pragma unroll
  for (int j = 0; j < 4; ++j) {
#pragma unroll
    for (int v = 0; v < 8; ++v) {
      const int m = m0 + wave * 16 + v + half * 8;   // C layout: M = vgpr + 8*(lane/16)
      const int n = n0 + j * 16 + lrow;              //           N = lane%16
      const float val = (acc[j][v] + bias[n]) * scale;
      if (outb) outb[(size_t)m * ldc + n] = (__bf16)val;
      else      outf[(size_t)m * ldc + n] = val;
    }
  }
}

// ---------------------------------------------------------------- attention
// QKV: bf16 [B*S][3072] rows = {q[h*64+e], k at +1024, v at +2048}, q/k/v pre-scaled 1/32.
// One block per (b, h, 128-query tile); 8 waves, wave owns 16 query rows.
// V tile is double-buffered: next tile's transpose-stage overlaps the score
// WMMAs + softmax of the current step. One barrier per key-step.
__global__ __launch_bounds__(256) void attn_kernel(const __bf16* __restrict__ QKV,
                                                   __bf16* __restrict__ Z)
{
  __shared__ __bf16 Vt[2][64 * 32];     // V^T tiles: [e][key], key contiguous
  __shared__ __bf16 Plds[8 * 16 * 32];  // per-wave P bounce (C-layout -> A-layout)
  const int t = threadIdx.x;
  const int wave = t >> 5, lane = t & 31;
  const int lrow = lane & 15, half = lane >> 4;
  const int qblk = blockIdx.x & 15;
  const int bh = blockIdx.x >> 4;
  const int b = bh >> 4, h = bh & 15;
  const __bf16* base = QKV + (size_t)b * SEQ * QKV_N;
  const int qr0 = qblk * 128 + wave * 16;

  // Q fragments (A-operand, 16x64 split into two 16x32): loaded once
  const __bf16* qp = base + (size_t)(qr0 + lrow) * QKV_N + h * D_HEAD + half * 16;
  const v16bf qlo = *(const v16bf*)qp;
  const v16bf qhi = *(const v16bf*)(qp + 32);

  // cooperative V transpose: thread t stages 8 elems of V[jb+key][e0..e0+7]
  const int vkey = t >> 3, ve0 = (t & 7) * 8;
  auto stageV = [&](int bufi, int jb) {
    const __bf16* vs = base + (size_t)(jb + vkey) * QKV_N + 2048 + h * D_HEAD + ve0;
    uint4 raw = *(const uint4*)vs;
    const __bf16* pv = (const __bf16*)&raw;
#pragma unroll
    for (int i = 0; i < 8; ++i) Vt[bufi][(ve0 + i) * 32 + vkey] = pv[i];
  };

  v8f o[4] = {};
  float mrun[8], lrun[8];
#pragma unroll
  for (int v = 0; v < 8; ++v) { mrun[v] = -1e30f; lrun[v] = 0.f; }

  const int kend = qblk * 128 + 128;
  stageV(0, 0);                 // prologue: first V tile into buffer 0
  int buf = 0;
  for (int jb = 0; jb < kend; jb += 32, buf ^= 1) {
    __syncthreads();                    // prev reads of buf^1 & writes of buf done
    if (jb + 32 < kend) stageV(buf ^ 1, jb + 32);   // overlaps score WMMAs/softmax

    // scores: two 16x16 tiles (keys jb..+15, jb+16..+31), K-dim = 64 in two steps
    const __bf16* kp0 = base + (size_t)(jb + lrow) * QKV_N + 1024 + h * D_HEAD + half * 16;
    const __bf16* kp1 = kp0 + (size_t)16 * QKV_N;
    const v16bf k00 = *(const v16bf*)kp0;
    const v16bf k01 = *(const v16bf*)(kp0 + 32);
    const v16bf k10 = *(const v16bf*)kp1;
    const v16bf k11 = *(const v16bf*)(kp1 + 32);
    v8f s0 = {}, s1 = {};
    s0 = WMMA_BF16(qlo, k00, s0);
    s0 = WMMA_BF16(qhi, k01, s0);
    s1 = WMMA_BF16(qlo, k10, s1);
    s1 = WMMA_BF16(qhi, k11, s1);

    float p0[8], p1[8];
#pragma unroll
    for (int v = 0; v < 8; ++v) {
      const int qg = qr0 + v + half * 8;               // this element's query row
      float a  = s0[v] * 0.125f;                       // remaining 1/sqrt(64)
      float bb = s1[v] * 0.125f;
      if (jb + lrow > qg)      a  = -1e30f;            // causal mask
      if (jb + 16 + lrow > qg) bb = -1e30f;
      float tmax = fmaxf(a, bb);
#pragma unroll
      for (int msk = 1; msk < 16; msk <<= 1)           // row-max across the 16 N-lanes
        tmax = fmaxf(tmax, __shfl_xor(tmax, msk, 32));
      const float nm = fmaxf(mrun[v], tmax);
      const float alpha = __expf(mrun[v] - nm);
      a = __expf(a - nm); bb = __expf(bb - nm);
      float rs = a + bb;
#pragma unroll
      for (int msk = 1; msk < 16; msk <<= 1)
        rs += __shfl_xor(rs, msk, 32);
      lrun[v] = lrun[v] * alpha + rs;
      mrun[v] = nm;
      p0[v] = a; p1[v] = bb;
      o[0][v] *= alpha; o[1][v] *= alpha; o[2][v] *= alpha; o[3][v] *= alpha;
    }

    // C-layout -> A-layout redistribution of P through per-wave LDS (no barrier:
    // wave-private region, DS ops are in-order within a wave)
    __bf16* pb = Plds + wave * 512;
#pragma unroll
    for (int v = 0; v < 8; ++v) {
      const int m = v + half * 8;
      pb[m * 32 + lrow]      = (__bf16)p0[v];
      pb[m * 32 + 16 + lrow] = (__bf16)p1[v];
    }
    const v16bf pfrag = *(const v16bf*)(pb + lrow * 32 + half * 16);

    // preload V fragments from the current buffer, then WMMAs back-to-back
    const __bf16* Vb = &Vt[buf][0];
    v16bf vfrag[4];
#pragma unroll
    for (int j = 0; j < 4; ++j)
      vfrag[j] = *(const v16bf*)(Vb + (j * 16 + lrow) * 32 + half * 16);
#pragma unroll
    for (int j = 0; j < 4; ++j)
      o[j] = WMMA_BF16(pfrag, vfrag[j], o[j]);
  }

  // normalize and write Z (bf16, [B*S][H*64])
#pragma unroll
  for (int j = 0; j < 4; ++j) {
#pragma unroll
    for (int v = 0; v < 8; ++v) {
      const int m = qr0 + v + half * 8;
      const float val = o[j][v] / lrun[v];
      Z[((size_t)b * SEQ + m) * D_MODEL + h * D_HEAD + j * 16 + lrow] = (__bf16)val;
    }
  }
}

// ---------------------------------------------------------------- launch
extern "C" void kernel_launch(void* const* d_in, const int* in_sizes, int n_in,
                              void* d_out, int out_size, void* d_ws, size_t ws_size,
                              hipStream_t stream)
{
  (void)in_sizes; (void)n_in; (void)out_size; (void)ws_size;
  const float* x  = (const float*)d_in[0];
  const float* wq = (const float*)d_in[1];
  const float* wk = (const float*)d_in[2];
  const float* wv = (const float*)d_in[3];
  const float* wo = (const float*)d_in[4];
  const float* bq = (const float*)d_in[5];
  const float* bk = (const float*)d_in[6];
  const float* bv = (const float*)d_in[7];
  const float* bo = (const float*)d_in[8];
  float* out = (float*)d_out;

  char* ws = (char*)d_ws;
  size_t off = 0;
  auto alloc = [&](size_t bytes) {
    void* p = ws + off;
    off = (off + bytes + 255) & ~(size_t)255;
    return p;
  };
  __bf16* X16   = (__bf16*)alloc((size_t)MROWS * D_MODEL * 2);   // 8 MB
  __bf16* Wqkvt = (__bf16*)alloc((size_t)QKV_N * D_MODEL * 2);   // 6 MB
  __bf16* WOt   = (__bf16*)alloc((size_t)D_MODEL * D_MODEL * 2); // 2 MB
  float*  biasq = (float*)alloc((size_t)QKV_N * 4);
  __bf16* QKVb  = (__bf16*)alloc((size_t)MROWS * QKV_N * 2);     // 24 MB
  __bf16* Z16   = (__bf16*)alloc((size_t)MROWS * D_MODEL * 2);   // 8 MB

  {
    int n = MROWS * D_MODEL;
    pack_x_kernel<<<(n + 255) / 256, 256, 0, stream>>>(x, X16, n);
  }
  pack_wqkv_kernel<<<(QKV_N * D_MODEL + 255) / 256, 256, 0, stream>>>(wq, wk, wv, Wqkvt);
  pack_wo_kernel<<<(D_MODEL * D_MODEL + 255) / 256, 256, 0, stream>>>(wo, WOt);
  pack_bias_kernel<<<(QKV_N + 255) / 256, 256, 0, stream>>>(bq, bk, bv, biasq);

  // q,k,v = (x@W + b) * 1/sqrt(1024)
  gemm_bf16_kernel<<<dim3(MROWS / 128, QKV_N / 64), 256, 0, stream>>>(
      X16, Wqkvt, biasq, 0.03125f, D_MODEL, QKV_N, QKVb, nullptr);

  attn_kernel<<<dim3(BATCH * N_HEADS * (SEQ / 128)), 256, 0, stream>>>(QKVb, Z16);

  // out = (z@W_O + b_O) * 1/sqrt(64)
  gemm_bf16_kernel<<<dim3(MROWS / 128, D_MODEL / 64), 256, 0, stream>>>(
      Z16, WOt, bo, 0.125f, D_MODEL, D_MODEL, nullptr, out);
}
